// ElmanMamba_7206955123460
// MI455X (gfx1250) — matrix operand order repaired and checked by reading, not compiled, not run on hardware
//
#include <hip/hip_runtime.h>
#include <cmath>

// Problem sizes (fixed by the reference)
#define TT   2048
#define BB   8
#define DD   1024
#define TOK  (TT * BB)          // 16384 tokens
#define REC_BLOCKS 8            // persistent recurrence workgroups (128 cols each)

typedef __bf16 bf16_t;
typedef __attribute__((ext_vector_type(16))) __bf16         v16bf;
typedef __attribute__((ext_vector_type(8)))  float          v8f32;
typedef __attribute__((ext_vector_type(8)))  unsigned short v8us;
typedef __attribute__((ext_vector_type(16))) unsigned short v16us;

// ---------------- helpers ----------------

__device__ __forceinline__ unsigned short f2bf_bits(float f) {
  unsigned u = __builtin_bit_cast(unsigned, f);
  u += 0x7FFFu + ((u >> 16) & 1u);          // round-to-nearest-even
  return (unsigned short)(u >> 16);
}

__device__ __forceinline__ v16bf frag16(v8us lo, v8us hi) {
  v16us w = __builtin_shufflevector(lo, hi, 0,1,2,3,4,5,6,7,8,9,10,11,12,13,14,15);
  return __builtin_bit_cast(v16bf, w);
}

__device__ __forceinline__ v16bf loadw32(const unsigned short* p) {
  // 16 consecutive bf16 (32B, 32B-aligned by construction)
  v16us w = *(const v16us*)p;
  return __builtin_bit_cast(v16bf, w);
}

__device__ __forceinline__ v8f32 wmma_bf16(v16bf a, v16bf b, v8f32 c) {
  return __builtin_amdgcn_wmma_f32_16x16x32_bf16(
      /*neg_a=*/false, a, /*neg_b=*/false, b,
      /*c_mod=*/(short)0, c, /*reuse_a=*/false, /*reuse_b=*/false);
}

__device__ __forceinline__ float fast_tanh(float x) {
#if __has_builtin(__builtin_amdgcn_tanhf)
  return __builtin_amdgcn_tanhf(x);          // native v_tanh_f32 (TRANS op)
#elif __has_builtin(__builtin_amdgcn_tanh_f32)
  return __builtin_amdgcn_tanh_f32(x);
#else
  return tanhf(x);
#endif
}

// Device-wide barrier (atomic generation counter). Cluster barrier is issued
// opportunistically first: it is architecturally a NOP when not launched as a
// cluster, and a fast SE-local sync if the runtime ever clusters this grid.
__device__ __forceinline__ void grid_sync(unsigned* cnt, unsigned* gen, unsigned nb) {
  __threadfence();                            // make this thread's stores agent-visible
  __syncthreads();
  if (threadIdx.x == 0) {
    if (__builtin_amdgcn_cluster_id_x() > 0)
      __builtin_amdgcn_s_cluster_barrier();
    unsigned g = __hip_atomic_load(gen, __ATOMIC_RELAXED, __HIP_MEMORY_SCOPE_AGENT);
    unsigned a = __hip_atomic_fetch_add(cnt, 1u, __ATOMIC_ACQ_REL, __HIP_MEMORY_SCOPE_AGENT);
    if (a == nb - 1u) {
      __hip_atomic_store(cnt, 0u, __ATOMIC_RELAXED, __HIP_MEMORY_SCOPE_AGENT);
      __hip_atomic_fetch_add(gen, 1u, __ATOMIC_RELEASE, __HIP_MEMORY_SCOPE_AGENT);
    } else {
      while (__hip_atomic_load(gen, __ATOMIC_ACQUIRE, __HIP_MEMORY_SCOPE_AGENT) == g)
        __builtin_amdgcn_s_sleep(1);
    }
  }
  __syncthreads();
}

// ---------------- kernels ----------------

// f32 -> bf16 bits (weights and x)
__global__ __launch_bounds__(256) void cvt_bf16_kernel(
    const float* __restrict__ src, unsigned short* __restrict__ dst, int n) {
  int i = blockIdx.x * blockDim.x + threadIdx.x;
  if (i < n) dst[i] = f2bf_bits(src[i]);
}

// h0 (f32 [8,1024]) -> bf16 rows 0..7 of h ping buffer (rest pre-zeroed)
__global__ __launch_bounds__(256) void h0init_kernel(
    const float* __restrict__ h0, unsigned short* __restrict__ hbuf) {
  int i = blockIdx.x * blockDim.x + threadIdx.x;   // 0..8191
  if (i < BB * DD) hbuf[i] = f2bf_bits(h0[i]);
}

// Fused input projections: for every token m and channel c
//   xwb   = x@Wx^T + bias
//   alpha = exp(-softplus(x@Wd^T + b_delta) * exp(-exp(A)))
//   gate  = silu(x@Wg^T + b_gate)
// One wave per 16x16 output tile, K=1024 in 32 bf16-WMMA steps; the bf16 A
// fragment (pre-converted x) is reused for all three B matrices.
__global__ __launch_bounds__(256) void proj_kernel(
    const unsigned short* __restrict__ xbf,
    const unsigned short* __restrict__ Wx,
    const unsigned short* __restrict__ Wd,
    const unsigned short* __restrict__ Wg,
    const float*          __restrict__ bias,
    const float*          __restrict__ bdelta,
    const float*          __restrict__ bgate,
    const float*          __restrict__ Avec,
    float* __restrict__ xwb, float* __restrict__ alpha, float* __restrict__ gate) {
  const int lane  = threadIdx.x & 31;
  const int wid   = blockIdx.x * (blockDim.x >> 5) + (threadIdx.x >> 5);
  const int mtile = wid >> 6;          // 0..1023  (token tiles)
  const int ntile = wid & 63;          // 0..63    (channel tiles)
  const int nn    = lane & 15;
  const int hh    = lane >> 4;         // K-half selector
  const int c     = ntile * 16 + nn;   // fixed channel per lane

  const float bias_c = bias[c];
  const float bd_c   = bdelta[c];
  const float bg_c   = bgate[c];
  const float decay  = __expf(-__expf(Avec[c]));

  const int m = lane & 15;             // A-matrix row within tile
  const unsigned short* xrow = xbf + (size_t)(mtile * 16 + m) * DD;
  const unsigned short* wxp  = Wx + (size_t)c * DD + 16 * hh;
  const unsigned short* wdp  = Wd + (size_t)c * DD + 16 * hh;
  const unsigned short* wgp  = Wg + (size_t)c * DD + 16 * hh;

  v8f32 accX = {}, accD = {}, accG = {};

#pragma unroll
  for (int kk = 0; kk < 32; ++kk) {
    const int kb = kk * 32;
    // A fragment: 16 consecutive K per lane half (matches B)
    v8us alo = *(const v8us*)(xrow + kb + 16 * hh);
    v8us ahi = *(const v8us*)(xrow + kb + 16 * hh + 8);
    v16bf afrag = frag16(alo, ahi);
    // B fragments: lane holds 16 consecutive K of weight-row c
    v16bf bx = loadw32(wxp + kb);
    v16bf bd = loadw32(wdp + kb);
    v16bf bg = loadw32(wgp + kb);
    accX = wmma_bf16(afrag, bx, accX);
    accD = wmma_bf16(afrag, bd, accD);
    accG = wmma_bf16(afrag, bg, accG);
  }

#pragma unroll
  for (int r = 0; r < 8; ++r) {
    const int token = mtile * 16 + r + 8 * hh;    // D layout: row = r + 8*half
    const size_t idx = (size_t)token * DD + c;
    float xv   = accX[r] + bias_c;
    float draw = accD[r] + bd_c;
    float sp   = (draw > 20.f) ? draw : log1pf(__expf(draw));
    float al   = __expf(-sp * decay);
    float g    = accG[r] + bg_c;
    float gt   = g / (1.f + __expf(-g));
    xwb[idx]   = xv;
    alpha[idx] = al;
    gate[idx]  = gt;
  }
}

// Persistent recurrence: 8 workgroups x 256 threads. Workgroup b owns output
// channels [128b, 128b+128); its 128 rows of R (bf16) are DMA'd into 256KB of
// LDS by the Tensor Data Mover (CDNA5 320KB WGP LDS). Per step: 32 chained
// bf16 WMMAs vs LDS-resident B, tanh blend in f32 registers (h-state lives in
// accumulator layout), publish bf16 h-slice to a double-buffered global h,
// device-wide barrier.
__global__ __launch_bounds__(256) void rec_kernel(
    const unsigned short* __restrict__ Rbf,
    const float* __restrict__ xwb,
    const float* __restrict__ alpha,
    const float* __restrict__ gate,
    const float* __restrict__ h0,
    unsigned short* __restrict__ hbuf,   // [2][16][1024] bf16, pre-zeroed
    float* __restrict__ out,
    unsigned* __restrict__ bar) {
  extern __shared__ unsigned short ldsR[];        // 128*1024 bf16 = 256 KB
  const int tid  = threadIdx.x;
  const int lane = tid & 31;
  const int wid  = tid >> 5;                      // 0..7: wave's 16-col tile
  const int nn   = lane & 15;
  const int hh   = lane >> 4;
  const int c    = blockIdx.x * 128 + wid * 16 + nn;

  // Stage this workgroup's R slice into LDS.
#if __has_builtin(__builtin_amdgcn_tensor_load_to_lds)
  {
    // This toolchain exposes the 6-argument form:
    //   (uint32x4 g0, int32x8 g1, int32x4 g2, int32x4 g3, int32x8 g4, i32 cpol)
    typedef __attribute__((ext_vector_type(4))) unsigned int u32x4;
    typedef __attribute__((ext_vector_type(8))) int          i32x8;
    typedef __attribute__((ext_vector_type(4))) int          i32x4;
    if (wid == 0) {
      // Tensor DMA descriptor (D#): 2D tile 1024x128 of 2-byte elements.
      unsigned long long ga =
          (unsigned long long)(const void*)(Rbf + (size_t)blockIdx.x * (128 * DD));
      unsigned ldsbase = (unsigned)__builtin_amdgcn_groupstaticsize();
      u32x4 g0;
      g0[0] = 1u;                                              // count=1 (user D#)
      g0[1] = ldsbase;                                         // lds_addr (bytes)
      g0[2] = (unsigned)ga;                                    // global_addr[31:0]
      g0[3] = (unsigned)((ga >> 32) & 0x01FFFFFFull) | (2u << 30); // [56:32] | type=2
      i32x8 g1;
      g1[0] = (int)(1u << 16);        // workgroup_mask=0, data_size=1 (2 bytes)
      g1[1] = (int)(1024u << 16);     // tensor_dim0[15:0]=1024 (bits 63:48)
      g1[2] = (int)(128u << 16);      // tensor_dim1[15:0]=128  (bits 95:80)
      g1[3] = (int)(1024u << 16);     // tile_dim0=1024         (bits 127:112)
      g1[4] = (int)128;               // tile_dim1=128          (bits 143:128)
      g1[5] = (int)1024;              // tensor_dim0_stride=1024 (bits 191:160)
      g1[6] = 0;                      // stride hi / tensor_dim1_stride lo
      g1[7] = 0;
      i32x4 g2 = {0, 0, 0, 0};        // dims 2/3 unused (2D tensor)
      i32x4 g3 = {0, 0, 0, 0};
      i32x8 g4 = {0, 0, 0, 0, 0, 0, 0, 0};
      __builtin_amdgcn_tensor_load_to_lds(g0, g1, g2, g3, g4, 0);
      __builtin_amdgcn_s_wait_tensorcnt(0);                    // TENSORcnt == 0
    }
    __syncthreads();
  }
#else
  {
    const v8us* src = (const v8us*)(Rbf + (size_t)blockIdx.x * (128 * DD));
    v8us* dst = (v8us*)ldsR;
#pragma unroll 4
    for (int i = tid; i < (128 * DD) / 8; i += 256) dst[i] = src[i];
    __syncthreads();
  }
#endif

  float hstate[8];
#pragma unroll
  for (int r = 0; r < 8; ++r) {
    int row = r + 8 * hh;
    hstate[r] = (row < BB) ? h0[(size_t)row * DD + c] : 0.f;
  }

  const int m = lane & 15;                         // A row (batch, padded to 16)
  unsigned* cnt = bar;
  unsigned* gen = bar + 1;
  const unsigned nb = gridDim.x;
  const unsigned short* ldsrow = ldsR + (size_t)(wid * 16 + nn) * DD + 16 * hh;

  for (int t = 0; t < TT; ++t) {
    const unsigned short* hcur  = hbuf + (size_t)(t & 1) * (16 * DD);
    unsigned short*       hnext = hbuf + (size_t)((t + 1) & 1) * (16 * DD);
    const unsigned short* hrow  = hcur + m * DD;

    v8f32 acc = {};
#pragma unroll
    for (int kk = 0; kk < 32; ++kk) {
      const int kb = kk * 32;
      v8us alo = *(const v8us*)(hrow + kb + 16 * hh);
      v8us ahi = *(const v8us*)(hrow + kb + 16 * hh + 8);
      v16bf afrag = frag16(alo, ahi);
      v16bf bfrag = loadw32(ldsrow + kb);          // ds_load from big LDS
      acc = wmma_bf16(afrag, bfrag, acc);
    }

    const size_t tb = (size_t)t * (BB * DD);
#pragma unroll
    for (int r = 0; r < 8; ++r) {
      const int row = r + 8 * hh;
      if (row < BB) {
        const size_t idx = tb + (size_t)row * DD + c;
        float cand = fast_tanh(acc[r] + xwb[idx]);
        float a    = alpha[idx];
        float hn   = a * hstate[r] + (1.f - a) * cand;
        hstate[r]  = hn;
        out[idx]   = hn * gate[idx];
        hnext[row * DD + c] = f2bf_bits(hn);
      }
    }
    // Prefetch next step's activations while waiting at the barrier.
    if (t + 1 < TT && nn == 0 && hh == 0) {
#pragma unroll
      for (int r = 0; r < 8; ++r) {
        const size_t nidx = tb + (size_t)(BB * DD) + (size_t)r * DD + c;
        __builtin_prefetch(xwb + nidx, 0, 1);
        __builtin_prefetch(alpha + nidx, 0, 1);
        __builtin_prefetch(gate + nidx, 0, 1);
      }
    }
    grid_sync(cnt, gen, nb);
  }
}

// ---------------- launch ----------------

extern "C" void kernel_launch(void* const* d_in, const int* in_sizes, int n_in,
                              void* d_out, int out_size, void* d_ws, size_t ws_size,
                              hipStream_t stream) {
  const float* x    = (const float*)d_in[0];   // [T,B,D]
  const float* h0   = (const float*)d_in[1];   // [B,D]
  const float* Wx   = (const float*)d_in[2];   // [D,D]
  const float* R    = (const float*)d_in[3];   // [D,D]
  const float* bias = (const float*)d_in[4];   // [D]
  const float* Wd   = (const float*)d_in[5];   // [D,D]
  const float* bd   = (const float*)d_in[6];   // [D]
  const float* A    = (const float*)d_in[7];   // [D]
  const float* Wg   = (const float*)d_in[8];   // [D,D]
  const float* bg   = (const float*)d_in[9];   // [D]
  float* out = (float*)d_out;                  // [T,B,D] f32

  char* ws = (char*)d_ws;
  const size_t W_ELEMS = (size_t)DD * DD;                    // 1M
  const size_t X_ELEMS = (size_t)TOK * DD;                   // 16.7M
  size_t off = 0;
  unsigned short* Wx_bf = (unsigned short*)(ws + off); off += W_ELEMS * 2;
  unsigned short* Wd_bf = (unsigned short*)(ws + off); off += W_ELEMS * 2;
  unsigned short* Wg_bf = (unsigned short*)(ws + off); off += W_ELEMS * 2;
  unsigned short* R_bf  = (unsigned short*)(ws + off); off += W_ELEMS * 2;
  unsigned short* x_bf  = (unsigned short*)(ws + off); off += X_ELEMS * 2;
  float* xwb   = (float*)(ws + off); off += X_ELEMS * 4;
  float* alpha = (float*)(ws + off); off += X_ELEMS * 4;
  float* gatep = (float*)(ws + off); off += X_ELEMS * 4;
  unsigned short* hbuf = (unsigned short*)(ws + off);
  const size_t HBUF_BYTES = (size_t)2 * 16 * DD * 2;         // 64 KB
  off += HBUF_BYTES;
  unsigned* bar = (unsigned*)(ws + off);
  const size_t BAR_BYTES = 256;
  (void)ws_size; (void)in_sizes; (void)n_in; (void)out_size;

  // 1) Convert weights and x to bf16 (B-fragment friendly row-major layout).
  cvt_bf16_kernel<<<4096, 256, 0, stream>>>(Wx, Wx_bf, (int)W_ELEMS);
  cvt_bf16_kernel<<<4096, 256, 0, stream>>>(Wd, Wd_bf, (int)W_ELEMS);
  cvt_bf16_kernel<<<4096, 256, 0, stream>>>(Wg, Wg_bf, (int)W_ELEMS);
  cvt_bf16_kernel<<<4096, 256, 0, stream>>>(R,  R_bf,  (int)W_ELEMS);
  cvt_bf16_kernel<<<65536, 256, 0, stream>>>(x, x_bf, (int)X_ELEMS);

  // 2) Zero h double-buffer (padded rows must be 0) + barrier words.
  (void)hipMemsetAsync((void*)hbuf, 0, HBUF_BYTES + BAR_BYTES, stream);
  h0init_kernel<<<(BB * DD + 255) / 256, 256, 0, stream>>>(h0, hbuf);

  // 3) Fused token-parallel projections (3 GEMMs + elementwise epilogue).
  //    65536 tiles of 16x16, 8 waves/block -> 8192 blocks.
  proj_kernel<<<8192, 256, 0, stream>>>(x_bf, Wx_bf, Wd_bf, Wg_bf,
                                        bias, bd, bg, A, xwb, alpha, gatep);

  // 4) Persistent sequential recurrence; R lives in 256KB LDS per WGP.
  (void)hipFuncSetAttribute((const void*)rec_kernel,
                            hipFuncAttributeMaxDynamicSharedMemorySize,
                            128 * DD * 2);
  rec_kernel<<<REC_BLOCKS, 256, 128 * DD * 2, stream>>>(
      R_bf, xwb, alpha, gatep, h0, hbuf, out, bar);
}
